// GuitarToPianoResynthModel_532575944792
// MI455X (gfx1250) — matrix operand order repaired and checked by reading, not compiled, not run on hardware
//
#include <hip/hip_runtime.h>
#include <math.h>

// ---------------------------------------------------------------------------
// CDNA5 (gfx1250) wave32 WMMA types
// ---------------------------------------------------------------------------
typedef __attribute__((ext_vector_type(16))) __bf16 v16bf;
typedef __attribute__((ext_vector_type(8)))  float  v8f;

union BfTile { v16bf v; __bf16 h[16]; unsigned int d[8]; };

#define ZACC {0.f,0.f,0.f,0.f,0.f,0.f,0.f,0.f}

__device__ __forceinline__ v8f wmma_bf16(const BfTile& a, const BfTile& b, v8f c) {
  return __builtin_amdgcn_wmma_f32_16x16x32_bf16(false, a.v, false, b.v,
                                                 (short)0, c, false, false);
}

__device__ __forceinline__ float geluf(float x) {
  return 0.5f * x * (1.0f + erff(x * 0.70710678118654752f));
}
__device__ __forceinline__ float sigm(float x) { return 1.0f / (1.0f + expf(-x)); }

// ---------------------------------------------------------------------------
// Weight packing: conv W[Co][Ci][Kw] (f32) -> bf16 [co][kw*Ci + ci]
// ---------------------------------------------------------------------------
__global__ void pack_conv_w_k(const float* __restrict__ W, __bf16* __restrict__ Wp,
                              int Co, int Ci, int Kw) {
  const int idx = blockIdx.x * 256 + threadIdx.x;
  const int n = Co * Ci * Kw;
  if (idx >= n) return;
  const int kw = idx % Kw;
  const int ci = (idx / Kw) % Ci;
  const int co = idx / (Kw * Ci);
  Wp[(size_t)co * Ci * Kw + (size_t)kw * Ci + ci] = (__bf16)W[idx];
}

__global__ void cvt_bf16_k(const float* __restrict__ s, __bf16* __restrict__ d, int n) {
  const int idx = blockIdx.x * 256 + threadIdx.x;
  if (idx < n) d[idx] = (__bf16)s[idx];
}

// ---------------------------------------------------------------------------
// in_conv (I=2,K=9) + GELU, direct (tiny K, not worth WMMA)
// ---------------------------------------------------------------------------
__global__ void in_conv_gelu_k(const float* __restrict__ xc, const float* __restrict__ oc,
                               const float* __restrict__ W, const float* __restrict__ bias,
                               float* __restrict__ y) {
  const int T = 12288;
  const int idx = blockIdx.x * 256 + threadIdx.x;
  if (idx >= 16 * 64 * T) return;
  const int t = idx % T;
  const int o = (idx / T) & 63;
  const int b = idx / (T * 64);
  const float* xcb = xc + (size_t)b * T;
  const float* ocb = oc + (size_t)b * T;
  float s = bias[o];
#pragma unroll
  for (int k = 0; k < 9; ++k) {
    const int tt = t - (8 - k);
    if (tt >= 0) {
      s += W[(o * 2 + 0) * 9 + k] * xcb[tt];
      s += W[(o * 2 + 1) * 9 + k] * ocb[tt];
    }
  }
  y[idx] = geluf(s);
}

// ---------------------------------------------------------------------------
// Causal dilated conv1d as implicit GEMM with bf16 WMMA, 2x2 register tiling.
// y[b][co][t] = bias[co] + sum_{ci,kw} W[co][ci][kw] * x[b][ci][t*stride-(Kw-1-kw)*dil]
// K ordering: k = kw*Ci + ci (Ci % 32 == 0 so each 32-chunk is single-tap).
// One wave = 32(co) x 32(t) output (4 accumulator tiles); 4 waves/block along t.
// A and B operands are each reused by two WMMAs -> ~12 loads per matrix op.
// ---------------------------------------------------------------------------
__global__ void __launch_bounds__(128) conv_wmma_k(
    const float* __restrict__ x, const __bf16* __restrict__ Wp,
    const float* __restrict__ bias, float* __restrict__ y,
    int Ci, int Ti, int Co, int To, int Kw, int stride, int dil, int fuse_gelu) {
  const int lane = threadIdx.x & 31;
  const int wv   = threadIdx.x >> 5;
  const int l15  = lane & 15;
  const int nh   = lane >> 4;
  const int t0   = (blockIdx.x * 4 + wv) * 32;  // wave covers 32 time steps
  const int co0  = blockIdx.y * 32;             // wave covers 32 out channels
  const int b    = blockIdx.z;
  const int Ktot = Ci * Kw;
  const float*  xb    = x + (size_t)b * Ci * Ti;
  const __bf16* wrow0 = Wp + (size_t)(co0 + l15) * Ktot;
  const __bf16* wrow1 = wrow0 + (size_t)16 * Ktot;

  v8f acc00 = ZACC, acc01 = ZACC, acc10 = ZACC, acc11 = ZACC;
  const int ta = t0 + l15;
  const int tbx = t0 + 16 + l15;
  for (int kw = 0; kw < Kw; ++kw) {
    const int off  = (Kw - 1 - kw) * dil;
    const int tina = ta * stride - off;
    const int tinb = tbx * stride - off;
    for (int cic = 0; cic < Ci; cic += 32) {
      const int kc = kw * Ci + cic;
      const int kh = kc + nh * 16;
      BfTile a0, a1, b0, b1;
      // A operands (weights): lane l<16 -> K {kc..kc+15}; l>=16 -> {kc+16..kc+31}
#pragma unroll
      for (int v2 = 0; v2 < 4; ++v2) {
        a0.d[v2]     = *(const unsigned int*)(wrow0 + kh + 2 * v2);
        a0.d[4 + v2] = *(const unsigned int*)(wrow0 + kh + 8 + 2 * v2);
        a1.d[v2]     = *(const unsigned int*)(wrow1 + kh + 2 * v2);
        a1.d[4 + v2] = *(const unsigned int*)(wrow1 + kh + 8 + 2 * v2);
      }
      // B operands (activations): lane holds column t, K = ci (coalesced over lanes)
      const float* xcol = xb + (size_t)(cic + nh * 16) * Ti;
      if (tina >= 0) {
#pragma unroll
        for (int j = 0; j < 16; ++j) b0.h[j] = (__bf16)xcol[(size_t)j * Ti + tina];
      } else {
#pragma unroll
        for (int j = 0; j < 16; ++j) b0.h[j] = (__bf16)0.f;
      }
      if (tinb >= 0) {
#pragma unroll
        for (int j = 0; j < 16; ++j) b1.h[j] = (__bf16)xcol[(size_t)j * Ti + tinb];
      } else {
#pragma unroll
        for (int j = 0; j < 16; ++j) b1.h[j] = (__bf16)0.f;
      }
      acc00 = wmma_bf16(a0, b0, acc00);
      acc01 = wmma_bf16(a0, b1, acc01);
      acc10 = wmma_bf16(a1, b0, acc10);
      acc11 = wmma_bf16(a1, b1, acc11);
    }
  }
#pragma unroll
  for (int r = 0; r < 8; ++r) {
    const int coA = co0 + r + nh * 8;
    const int coB = coA + 16;
    float v00 = acc00[r] + bias[coA];
    float v01 = acc01[r] + bias[coA];
    float v10 = acc10[r] + bias[coB];
    float v11 = acc11[r] + bias[coB];
    if (fuse_gelu) { v00 = geluf(v00); v01 = geluf(v01); v10 = geluf(v10); v11 = geluf(v11); }
    y[((size_t)b * Co + coA) * To + ta]  = v00;
    y[((size_t)b * Co + coA) * To + tbx] = v01;
    y[((size_t)b * Co + coB) * To + ta]  = v10;
    y[((size_t)b * Co + coB) * To + tbx] = v11;
  }
}

// ---------------------------------------------------------------------------
// GroupNorm(8) + optional GELU + optional residual add, in-place capable.
// One block per (group, batch).
// ---------------------------------------------------------------------------
__global__ void __launch_bounds__(256) gnorm_k(
    const float* __restrict__ x, const float* __restrict__ sc, const float* __restrict__ sh,
    float* __restrict__ out, const float* __restrict__ res, int C, int T, int do_gelu) {
  __shared__ float r0[256], r1[256];
  const int g = blockIdx.x, b = blockIdx.y;
  const int cg = C >> 3;
  const size_t base = ((size_t)b * C + (size_t)g * cg) * T;
  const int n = cg * T;
  float s0 = 0.f, s1 = 0.f;
  for (int i = threadIdx.x; i < n; i += 256) { float v = x[base + i]; s0 += v; s1 += v * v; }
  r0[threadIdx.x] = s0; r1[threadIdx.x] = s1;
  __syncthreads();
  for (int off = 128; off > 0; off >>= 1) {
    if ((int)threadIdx.x < off) { r0[threadIdx.x] += r0[threadIdx.x + off]; r1[threadIdx.x] += r1[threadIdx.x + off]; }
    __syncthreads();
  }
  const float m    = r0[0] / (float)n;
  const float var  = fmaxf(r1[0] / (float)n - m * m, 0.f);
  const float rinv = rsqrtf(var + 1e-5f);
  for (int i = threadIdx.x; i < n; i += 256) {
    const int cl = g * cg + i / T;
    float v = (x[base + i] - m) * rinv * sc[cl] + sh[cl];
    if (do_gelu) v = geluf(v);
    if (res) v += res[base + i];
    out[base + i] = v;
  }
}

// ---------------------------------------------------------------------------
// GRU input projection GEMM: gi[t][b][n] = sum_k X(b,k,t)*Wih[n][k] + bih[n]
// Rows m = t*16+b; K=256 (8 chunks); N=768. One wave = 2 N-tiles sharing one
// A gather (the expensive strided f32 activation read).
// X addressed generically: X[b*sb + k*sk + t*st].
// ---------------------------------------------------------------------------
__global__ void __launch_bounds__(128) gi_gemm_k(
    const float* __restrict__ X, const __bf16* __restrict__ Wb,
    const float* __restrict__ bias, float* __restrict__ gi,
    long long sb, long long sk, long long st) {
  const int lane = threadIdx.x & 31;
  const int wv   = threadIdx.x >> 5;
  const int l15  = lane & 15;
  const int nh   = lane >> 4;
  const int n0   = (blockIdx.x * 4 + wv) * 32;  // two 16-wide N tiles
  const int t    = blockIdx.y;

  v8f acc0 = ZACC, acc1 = ZACC;
  for (int kc = 0; kc < 256; kc += 32) {
    BfTile a, bt0, bt1;
    const long long boff = (long long)l15 * sb + (long long)t * st;
#pragma unroll
    for (int v2 = 0; v2 < 8; ++v2) {
      const int k0 = kc + nh * 16 + 2 * v2;
      a.h[2 * v2]     = (__bf16)X[(size_t)(boff + (long long)k0 * sk)];
      a.h[2 * v2 + 1] = (__bf16)X[(size_t)(boff + (long long)(k0 + 1) * sk)];
    }
    const __bf16* wrow0 = Wb + (size_t)(n0 + l15) * 256 + kc + nh * 16;
    const __bf16* wrow1 = wrow0 + (size_t)16 * 256;
#pragma unroll
    for (int v2 = 0; v2 < 8; ++v2) {
      bt0.d[v2] = *(const unsigned int*)(wrow0 + 2 * v2);
      bt1.d[v2] = *(const unsigned int*)(wrow1 + 2 * v2);
    }
    acc0 = wmma_bf16(a, bt0, acc0);
    acc1 = wmma_bf16(a, bt1, acc1);
  }
  const float bi0 = bias[n0 + l15];
  const float bi1 = bias[n0 + 16 + l15];
#pragma unroll
  for (int r = 0; r < 8; ++r) {
    const int b = r + nh * 8;
    gi[((size_t)t * 16 + b) * 768 + n0 + l15]      = acc0[r] + bi0;
    gi[((size_t)t * 16 + b) * 768 + n0 + 16 + l15] = acc1[r] + bi1;
  }
}

// ---------------------------------------------------------------------------
// GRU recurrent scan: single persistent block, 16 waves.
// Wave w owns columns j = w*16 + l15; computes r/z/n WMMA tiles for its own
// columns so the gate update is wave-local. h kept in LDS (f32 + bf16 mirror).
// Next step's gi row is prefetched (global_prefetch_b8) behind the WMMAs.
// ---------------------------------------------------------------------------
__global__ void __launch_bounds__(512) gru_scan_k(
    const float* __restrict__ gi, const __bf16* __restrict__ Whh,
    const float* __restrict__ bhh, float* __restrict__ ys, int Tlen) {
  __shared__ __bf16 h_bf[16][264];
  __shared__ float  h_f[16][257];
  const int tid  = threadIdx.x;
  const int lane = tid & 31;
  const int w    = tid >> 5;
  const int l15  = lane & 15;
  const int nh   = lane >> 4;

  for (int i = tid; i < 16 * 264; i += 512) ((__bf16*)h_bf)[i] = (__bf16)0.f;
  for (int i = tid; i < 16 * 257; i += 512) ((float*)h_f)[i] = 0.f;
  __syncthreads();

  const int j = w * 16 + l15;
  const float bh_r = bhh[j], bh_z = bhh[j + 256], bh_n = bhh[j + 512];
  const __bf16* Wr = Whh + (size_t)j * 256;
  const __bf16* Wz = Whh + (size_t)(j + 256) * 256;
  const __bf16* Wn = Whh + (size_t)(j + 512) * 256;

  for (int t = 0; t < Tlen; ++t) {
    v8f ar_ = ZACC, az_ = ZACC, an_ = ZACC;
    for (int kc = 0; kc < 256; kc += 32) {
      BfTile a, br_, bz_, bn_;
      const int kh = kc + nh * 16;
#pragma unroll
      for (int v2 = 0; v2 < 4; ++v2) {
        a.d[v2]     = *(const unsigned int*)&h_bf[l15][kh + 2 * v2];
        a.d[4 + v2] = *(const unsigned int*)&h_bf[l15][kh + 8 + 2 * v2];
      }
      const int kb = kc + nh * 16;
#pragma unroll
      for (int v2 = 0; v2 < 8; ++v2) {
        br_.d[v2] = *(const unsigned int*)(Wr + kb + 2 * v2);
        bz_.d[v2] = *(const unsigned int*)(Wz + kb + 2 * v2);
        bn_.d[v2] = *(const unsigned int*)(Wn + kb + 2 * v2);
      }
      ar_ = wmma_bf16(a, br_, ar_);
      az_ = wmma_bf16(a, bz_, az_);
      an_ = wmma_bf16(a, bn_, an_);
    }
    // pull next step's gi row toward the caches while WMMAs drain
    if (t + 1 < Tlen)
      __builtin_prefetch(gi + (size_t)(t + 1) * 12288 + (size_t)tid * 24, 0, 3);
    __syncthreads();  // all h_bf reads for step t complete
    const float* git = gi + (size_t)t * 16 * 768;
#pragma unroll
    for (int r = 0; r < 8; ++r) {
      const int b = r + nh * 8;
      const float gr = git[b * 768 + j]       + ar_[r] + bh_r;
      const float gz = git[b * 768 + j + 256] + az_[r] + bh_z;
      const float gn = git[b * 768 + j + 512];
      const float hn = an_[r] + bh_n;
      const float rr = sigm(gr);
      const float zz = sigm(gz);
      const float nn = tanhf(gn + rr * hn);
      const float hp = h_f[b][j];
      const float hN = (1.f - zz) * nn + zz * hp;
      h_f[b][j]  = hN;
      h_bf[b][j] = (__bf16)hN;
      ys[((size_t)t * 16 + b) * 256 + j] = hN;
    }
    __syncthreads();  // h updated before next step's reads
  }
}

// ---------------------------------------------------------------------------
// Heads: hout[head][b][o]; head 0 = note_logits raw, 1=vel 2=bright 3=noise (sigmoid)
// ---------------------------------------------------------------------------
__global__ void heads_k(const float* __restrict__ feat,
                        const float* Wn, const float* bn,
                        const float* Wv, const float* bv,
                        const float* Wbr, const float* bbr,
                        const float* Wno, const float* bno,
                        float* __restrict__ hout) {
  const int idx = blockIdx.x * 256 + threadIdx.x;
  if (idx >= 4 * 16 * 88) return;
  const int head = idx / (16 * 88);
  const int rem  = idx % (16 * 88);
  const int b = rem / 88, o = rem % 88;
  const float *W, *bb_;
  if (head == 0)      { W = Wn;  bb_ = bn;  }
  else if (head == 1) { W = Wv;  bb_ = bv;  }
  else if (head == 2) { W = Wbr; bb_ = bbr; }
  else                { W = Wno; bb_ = bno; }
  float s = bb_[o];
  const float* f  = feat + (size_t)b * 256;
  const float* wr = W + (size_t)o * 256;
  for (int k = 0; k < 256; ++k) s += f[k] * wr[k];
  if (head > 0) s = sigm(s);
  hout[idx] = s;
}

// ---------------------------------------------------------------------------
// Synth: closed-form phase (phase_s = (s+1)*inc); decays as pow; threefry noise.
// noisy = sum_n c[n]*white[n] with analytic kernel-column weights c[n].
// ---------------------------------------------------------------------------
__device__ __forceinline__ void threefry2x32(unsigned k0, unsigned k1,
                                             unsigned c0, unsigned c1,
                                             unsigned& o0, unsigned& o1) {
  const unsigned ks2 = 0x1BD11BDAu ^ k0 ^ k1;
  unsigned x0 = c0 + k0, x1 = c1 + k1;
#define TFR(r) { x0 += x1; x1 = (x1 << r) | (x1 >> (32 - r)); x1 ^= x0; }
  TFR(13) TFR(15) TFR(26) TFR(6)  x0 += k1;  x1 += ks2 + 1u;
  TFR(17) TFR(29) TFR(16) TFR(24) x0 += ks2; x1 += k0 + 2u;
  TFR(13) TFR(15) TFR(26) TFR(6)  x0 += k0;  x1 += k1 + 3u;
  TFR(17) TFR(29) TFR(16) TFR(24) x0 += k1;  x1 += ks2 + 4u;
  TFR(13) TFR(15) TFR(26) TFR(6)  x0 += ks2; x1 += k0 + 5u;
#undef TFR
  o0 = x0; o1 = x1;
}

__global__ void synth_k(const float* __restrict__ hout,
                        const float* __restrict__ decay_l, const float* __restrict__ harm_l,
                        const float* __restrict__ mg_p, const float* __restrict__ ndec_l,
                        const float* __restrict__ ng_p, float* __restrict__ out) {
  const int b = blockIdx.x;
  const int tid = threadIdx.x;
  __shared__ float s_amp[88], s_ns[88], s_nd[88], s_nsd[88], s_f0[88], s_h[88][8];
  if (tid < 88) {
    const int o = tid;
    const float nl  = hout[0 * 1408 + b * 88 + o];
    const float vel = hout[1 * 1408 + b * 88 + o];
    const float br  = hout[2 * 1408 + b * 88 + o];
    const float na  = hout[3 * 1408 + b * 88 + o];
    const float gate = sigm(nl);
    s_amp[o] = gate * vel;                 // max(0*decay, gate*vel)
    s_ns[o]  = gate * na;                  // relu(gate)*noise_amt
    s_nd[o]  = 0.9995f + 0.00049f * sigm(decay_l[o]);
    s_nsd[o] = 0.9f + 0.09f * sigm(ndec_l[o]);
    float hl[8], mx = -1e30f;
    for (int p = 0; p < 8; ++p) { hl[p] = harm_l[o * 8 + p]; mx = fmaxf(mx, hl[p]); }
    float se = 0.f;
    for (int p = 0; p < 8; ++p) { hl[p] = expf(hl[p] - mx); se += hl[p]; }
    float hs = 0.f;
    for (int p = 0; p < 8; ++p) {
      const float tv = (hl[p] / se) * expf(-(float)(p + 1) * (1.5f - 1.2f * br));
      hl[p] = tv; hs += tv;
    }
    const float inv = 1.f / (hs + 1e-8f);
    for (int p = 0; p < 8; ++p) s_h[o][p] = hl[p] * inv;
    s_f0[o] = 440.f * exp2f(((21.f + (float)o) - 69.f) / 12.f);
  }
  __syncthreads();
  if (tid < 96) {
    const int s = tid;
    const float mg = mg_p[0], ng = ng_p[0];
    float tonal = 0.f, noisy = 0.f;
    for (int o = 0; o < 88; ++o) {
      const float a   = s_amp[o] * powf(s_nd[o], (float)s);
      const float nsv = s_ns[o]  * powf(s_nsd[o], (float)s);
      const float base = 6.283185307179586f * s_f0[o] * (float)(s + 1) / 48000.f;
      float ss = 0.f;
#pragma unroll
      for (int p = 1; p <= 8; ++p) ss += sinf(base * (float)p) * s_h[o][p - 1];
      tonal += ss * a;
      const unsigned id = ((unsigned)(s * 16 + b)) * 88u + (unsigned)o;
      unsigned r0, r1;
      threefry2x32(0u, 123u, id, 0x9E3779B9u, r0, r1);
      const float u1 = fmaxf((float)(r0 >> 8) * (1.f / 16777216.f), 1e-7f);
      const float u2 = (float)(r1 >> 8) * (1.f / 16777216.f);
      const float nrm = sqrtf(-2.f * logf(u1)) * cosf(6.283185307179586f * u2);
      float cw = 1.f;
      if (o == 0) cw = 0.55f; else if (o == 1) cw = 0.90f;
      else if (o == 86) cw = 0.95f; else if (o == 87) cw = 0.80f;
      noisy += cw * nrm * nsv;
    }
    out[b * 96 + s] = mg * tonal + ng * noisy;
  }
}

// ---------------------------------------------------------------------------
// Host orchestration
// ---------------------------------------------------------------------------
extern "C" void kernel_launch(void* const* d_in, const int* in_sizes, int n_in,
                              void* d_out, int out_size, void* d_ws, size_t ws_size,
                              hipStream_t stream) {
  (void)in_sizes; (void)n_in; (void)out_size; (void)ws_size;
  // Leaf order (jax tree flatten, dicts sorted by key):
  // cnn.e1 [0..25], cnn.e2 [26..51], cnn.e3 [52..77], cnn.in_conv [78,79],
  // cnn.mid [80..111], gru.l0 [112..115], gru.l1 [116..119],
  // heads: bright[120,121] noise[122,123] note[124,125] vel[126,127],
  // onset_ctx[128], synth: decay[129] harm[130] mg[131] ndecay[132] ng[133], x_ctx[134]
  auto IN = [&](int i) { return (const float*)d_in[i]; };

  char* ws = (char*)d_ws;
  const size_t OFF_A  = 0;
  const size_t OFF_B  = 50331648;   // 16*786432 f32
  const size_t OFF_C  = 100663296;
  const size_t OFF_W  = 150994944;  // packed bf16 weights (17,154,048 B)
  const size_t OFF_YS = 168148992;  // 1536*16*256 f32
  const size_t OFF_HO = 193314816;  // head outputs
  float*  A    = (float*)(ws + OFF_A);
  float*  Bb   = (float*)(ws + OFF_B);
  float*  C    = (float*)(ws + OFF_C);
  float*  gi   = (float*)(ws + OFF_B);  // overlaps B+C, used only after CNN
  __bf16* wb   = (__bf16*)(ws + OFF_W);
  float*  ys   = (float*)(ws + OFF_YS);
  float*  hout = (float*)(ws + OFF_HO);

  size_t woff = 0;
  auto packw = [&](int widx, int Co, int Ci, int Kw) -> const __bf16* {
    __bf16* dst = wb + woff;
    const int n = Co * Ci * Kw;
    pack_conv_w_k<<<(n + 255) / 256, 256, 0, stream>>>(IN(widx), dst, Co, Ci, Kw);
    woff += (size_t)n;
    return dst;
  };
  auto packcvt = [&](int widx, int n) -> const __bf16* {
    __bf16* dst = wb + woff;
    cvt_bf16_k<<<(n + 255) / 256, 256, 0, stream>>>(IN(widx), dst, n);
    woff += (size_t)n;
    return dst;
  };
  auto convk = [&](const float* x, const __bf16* Wp, int bidx, float* y,
                   int Ci, int Ti, int Co, int stride, int dil, int Kw, int fg) {
    const int To = Ti / stride;
    conv_wmma_k<<<dim3(To / 128, Co / 32, 16), 128, 0, stream>>>(
        x, Wp, IN(bidx), y, Ci, Ti, Co, To, Kw, stride, dil, fg);
  };
  auto gnormk = [&](float* x, int sidx, int bidx, int Cc, int T, const float* res) {
    gnorm_k<<<dim3(8, 16), 256, 0, stream>>>(x, IN(sidx), IN(bidx), x, res, Cc, T, 1);
  };
  auto resblock = [&](float* xin, float* tmp, float* yout, int base, int Cc, int T, int dil) {
    convk(xin, packw(base + 0, Cc, Cc, 5), base + 1, tmp, Cc, T, Cc, 1, dil, 5, 0);
    gnormk(tmp, base + 4, base + 5, Cc, T, nullptr);
    convk(tmp, packw(base + 2, Cc, Cc, 5), base + 3, yout, Cc, T, Cc, 1, 1, 5, 0);
    gnormk(yout, base + 6, base + 7, Cc, T, xin);  // out = x + gelu(gnorm(conv2))
  };

  // ---- in_conv + gelu -> A [16,64,12288] ----
  in_conv_gelu_k<<<(16 * 64 * 12288 + 255) / 256, 256, 0, stream>>>(
      IN(134), IN(128), IN(78), IN(79), A);

  // ---- e1 ----
  convk(A, packw(0, 128, 64, 7), 1, Bb, 64, 12288, 128, 2, 1, 7, 1);
  resblock(Bb, C, A, 2, 128, 6144, 1);
  resblock(A, C, Bb, 10, 128, 6144, 2);
  resblock(Bb, C, A, 18, 128, 6144, 4);
  // ---- e2 ----
  convk(A, packw(26, 256, 128, 7), 27, Bb, 128, 6144, 256, 2, 1, 7, 1);
  resblock(Bb, C, A, 28, 256, 3072, 1);
  resblock(A, C, Bb, 36, 256, 3072, 2);
  resblock(Bb, C, A, 44, 256, 3072, 4);
  // ---- e3 ----
  convk(A, packw(52, 256, 256, 7), 53, Bb, 256, 3072, 256, 2, 1, 7, 1);
  resblock(Bb, C, A, 54, 256, 1536, 1);
  resblock(A, C, Bb, 62, 256, 1536, 2);
  resblock(Bb, C, A, 70, 256, 1536, 4);
  // ---- mid (x ends in A) ----
  resblock(A, C, Bb, 80, 256, 1536, 1);
  resblock(Bb, C, A, 88, 256, 1536, 2);
  resblock(A, C, Bb, 96, 256, 1536, 4);
  resblock(Bb, C, A, 104, 256, 1536, 8);

  // ---- GRU ----
  const __bf16* wih0 = packcvt(112, 768 * 256);
  const __bf16* whh0 = packcvt(113, 768 * 256);
  const __bf16* wih1 = packcvt(116, 768 * 256);
  const __bf16* whh1 = packcvt(117, 768 * 256);

  // layer 0: X = z in A, layout [b][k][t]: sb=256*1536, sk=1536, st=1
  gi_gemm_k<<<dim3(6, 1536), 128, 0, stream>>>(A, wih0, IN(114), gi,
                                               (long long)256 * 1536, 1536LL, 1LL);
  gru_scan_k<<<1, 512, 0, stream>>>(gi, whh0, IN(115), ys, 1536);
  // layer 1: X = ys, layout [t][b][k]: sb=256, sk=1, st=4096
  gi_gemm_k<<<dim3(6, 1536), 128, 0, stream>>>(ys, wih1, IN(118), gi,
                                               256LL, 1LL, 4096LL);
  gru_scan_k<<<1, 512, 0, stream>>>(gi, whh1, IN(119), ys, 1536);

  // ---- heads: feat = ys[T-1] ----
  const float* feat = ys + (size_t)1535 * 16 * 256;
  heads_k<<<(4 * 16 * 88 + 255) / 256, 256, 0, stream>>>(
      feat, IN(124), IN(125), IN(126), IN(127), IN(120), IN(121), IN(122), IN(123), hout);

  // ---- synth -> d_out [16,1,96] ----
  synth_k<<<16, 96, 0, stream>>>(hout, IN(129), IN(130), IN(131), IN(132), IN(133),
                                 (float*)d_out);
}
